// SemanticAlignmenCalibration_33612414058589
// MI455X (gfx1250) — compile-verified, matrix operand order checked by reading, not run on hardware
//
#include <hip/hip_runtime.h>
#include <hip/hip_bf16.h>
#include <math.h>

// ---------------------------------------------------------------------------
// CDNA5 (gfx1250, wave32) implementation of SemanticAlignmentCalibration.
// All conv GEMMs run on v_wmma_f32_16x16x32_bf16 (f32 accumulate).
// GEMM mapping: A = weights (M = out-channels), B = activations (N = spatial),
// so D stores are coalesced along x for NCHW outputs.
// ---------------------------------------------------------------------------

typedef __attribute__((ext_vector_type(16))) __bf16 v16bf;
typedef __attribute__((ext_vector_type(8)))  float  v8f;

#define BN_EPS 1e-3f

__device__ __forceinline__ unsigned bf16r(float f) {
  unsigned u = __float_as_uint(f);
  return (u + 0x7FFFu + ((u >> 16) & 1u)) >> 16;   // round-to-nearest-even
}
__device__ __forceinline__ unsigned pack2bf(float a, float b) {
  return bf16r(a) | (bf16r(b) << 16);
}

union Frag {
  uint4 q[2];
  v16bf v;
};

// LDS row layout: 16 uints = 32 bf16, K = 0..31 in order.
// Per the 16-bit A/B WMMA layout: lane(half = lane>>4) reads
// K runs [8*half, 8*half+8) and [16+8*half, 16+8*half+8)  => uints
// [4*half, 4*half+4) and [8+4*half, 12+4*half): two ds_load_b128.
__device__ __forceinline__ void load_frag(const unsigned* row, int half, Frag& f) {
  f.q[0] = *reinterpret_cast<const uint4*>(row + 4 * half);
  f.q[1] = *reinterpret_cast<const uint4*>(row + 8 + 4 * half);
}

__device__ __forceinline__ v8f wmma_bf16(const v16bf& a, const v16bf& b, v8f c) {
  return __builtin_amdgcn_wmma_f32_16x16x32_bf16(
      false, a, false, b, (short)0, c, false, false);
}

__device__ __forceinline__ v8f vzero8() {
  v8f z;
#pragma unroll
  for (int i = 0; i < 8; ++i) z[i] = 0.f;
  return z;
}

__device__ __forceinline__ float silu(float t) {
  return t / (1.f + __expf(-t));
}

// ---------------------------------------------------------------------------
// Generic 3x3 conv (pad=1) as implicit GEMM.  EPI: 0 = raw, 1 = BN+SiLU.
// Workgroup: 256 threads = 8 waves; tile = 128 spatial x CO channels.
// K loop: 9 taps x CIN/32 chunks of 32 bf16 K values.
// ---------------------------------------------------------------------------
template <int CIN, int CO, int CO_REAL, int HH, int WW, int EPI>
__global__ __launch_bounds__(256) void conv3x3_wmma(
    const float* __restrict__ in, const float* __restrict__ w,
    const float* __restrict__ bnp, float* __restrict__ out) {
  static_assert(CO % 16 == 0 && CIN % 32 == 0, "tile shape");
  __shared__ unsigned lds_a[128][16];   // activations [spatial][K]
  __shared__ unsigned lds_w[CO][16];    // weights     [cout][K]
  __shared__ float bninv[CO], bnsh[CO];

  const int tid = threadIdx.x;
  const int lane = tid & 31, wave = tid >> 5;
  const int half = lane >> 4, lanelo = lane & 15;
  const int p0 = blockIdx.x * 128;
  const int HWp = HH * WW;

  __builtin_prefetch((const void*)w, 0, 1);   // global_prefetch_b8

  if (EPI == 1 && tid < CO_REAL) {
    float g  = bnp[tid];
    float bb = bnp[CO_REAL + tid];
    float m  = bnp[2 * CO_REAL + tid];
    float vv = bnp[3 * CO_REAL + tid];
    float inv = g * rsqrtf(vv + BN_EPS);
    bninv[tid] = inv;
    bnsh[tid]  = bb - m * inv;
  }

  v8f acc[CO / 16];
#pragma unroll
  for (int i = 0; i < CO / 16; ++i) acc[i] = vzero8();

  for (int tap = 0; tap < 9; ++tap) {
    const int ky = tap / 3 - 1, kx = tap % 3 - 1;
    for (int cc0 = 0; cc0 < CIN; cc0 += 32) {
      __syncthreads();
      // ---- stage activation tile (im2col gather, f32 -> bf16) ----
      for (int u = tid; u < 128 * 16; u += 256) {
        int ml = u & 127, ku = u >> 7;          // lanes sweep spatial: coalesced
        int ci = cc0 + 2 * ku;
        int p = p0 + ml;
        int b = p / HWp, r = p % HWp;
        int y = r / WW + ky, x = r % WW + kx;
        float f0 = 0.f, f1 = 0.f;
        if ((unsigned)y < (unsigned)HH && (unsigned)x < (unsigned)WW) {
          const float* base = in + (((long)b * CIN + ci) * HH + y) * WW + x;
          f0 = base[0];
          f1 = base[HWp];
        }
        lds_a[ml][ku] = pack2bf(f0, f1);
      }
      // ---- stage weight tile ----
      for (int u = tid; u < CO * 16; u += 256) {
        int n = u % CO, ku = u / CO;
        int ci = cc0 + 2 * ku;
        float f0 = 0.f, f1 = 0.f;
        if (n < CO_REAL) {
          const float* wp = w + ((long)n * CIN + ci) * 9 + tap;
          f0 = wp[0];
          f1 = wp[9];
        }
        lds_w[n][ku] = pack2bf(f0, f1);
      }
      __syncthreads();
      // ---- WMMA: A=weights, B=activations ----
      Frag bf;
      load_frag(lds_a[wave * 16 + lanelo], half, bf);
#pragma unroll
      for (int mb = 0; mb < CO / 16; ++mb) {
        Frag af;
        load_frag(lds_w[mb * 16 + lanelo], half, af);
        acc[mb] = wmma_bf16(af.v, bf.v, acc[mb]);
      }
    }
  }

  // ---- epilogue: D lane layout N=lane&15 (spatial), M=r+8*half (channel) ----
  int p = p0 + wave * 16 + lanelo;
  int b = p / HWp, r = p % HWp;
  int y = r / WW, x = r % WW;
#pragma unroll
  for (int mb = 0; mb < CO / 16; ++mb) {
#pragma unroll
    for (int rr = 0; rr < 8; ++rr) {
      int c = mb * 16 + rr + 8 * half;
      if (c < CO_REAL) {
        float v = acc[mb][rr];
        if (EPI == 1) {
          v = v * bninv[c] + bnsh[c];
          v = silu(v);
        }
        out[(((long)b * CO_REAL + c) * HH + y) * WW + x] = v;
      }
    }
  }
}

// ---------------------------------------------------------------------------
// Fused 1x1 convs on sem: x1, x2, gate-logit (3 GEMMs sharing the B read).
// K=64, CO=64 each, spatial = 8*128*128 = 131072.
// ---------------------------------------------------------------------------
__global__ __launch_bounds__(256) void ffm_pre_wmma(
    const float* __restrict__ sem,
    const float* __restrict__ w1, const float* __restrict__ b1,
    const float* __restrict__ w2, const float* __restrict__ b2,
    const float* __restrict__ wg, const float* __restrict__ bg,
    float* __restrict__ x1, float* __restrict__ x2, float* __restrict__ gl) {
  __shared__ unsigned lds_a[128][16];
  __shared__ unsigned lds_w[3][64][16];
  const int tid = threadIdx.x;
  const int lane = tid & 31, wave = tid >> 5;
  const int half = lane >> 4, lanelo = lane & 15;
  const int p0 = blockIdx.x * 128;
  const float* ws[3] = {w1, w2, wg};
  const float* bs[3] = {b1, b2, bg};
  float* outs[3] = {x1, x2, gl};

  v8f acc[3][4];
#pragma unroll
  for (int s = 0; s < 3; ++s)
#pragma unroll
    for (int m = 0; m < 4; ++m) acc[s][m] = vzero8();

  for (int cc0 = 0; cc0 < 64; cc0 += 32) {
    __syncthreads();
    for (int u = tid; u < 128 * 16; u += 256) {
      int ml = u & 127, ku = u >> 7;
      int ci = cc0 + 2 * ku;
      int p = p0 + ml;
      int b = p >> 14, pix = p & 16383;
      const float* base = sem + (((long)b * 64 + ci) << 14) + pix;
      lds_a[ml][ku] = pack2bf(base[0], base[16384]);
    }
    for (int u = tid; u < 3 * 64 * 16; u += 256) {
      int set = u >> 10, rem = u & 1023;
      int n = rem & 63, ku = rem >> 6;
      int ci = cc0 + 2 * ku;
      const float* wp = ws[set] + n * 64 + ci;
      lds_w[set][n][ku] = pack2bf(wp[0], wp[1]);
    }
    __syncthreads();
    Frag bf;
    load_frag(lds_a[wave * 16 + lanelo], half, bf);
#pragma unroll
    for (int s = 0; s < 3; ++s)
#pragma unroll
      for (int mb = 0; mb < 4; ++mb) {
        Frag af;
        load_frag(lds_w[s][mb * 16 + lanelo], half, af);
        acc[s][mb] = wmma_bf16(af.v, bf.v, acc[s][mb]);
      }
  }

  int p = p0 + wave * 16 + lanelo;
  int b = p >> 14, pix = p & 16383;
#pragma unroll
  for (int s = 0; s < 3; ++s)
#pragma unroll
    for (int mb = 0; mb < 4; ++mb)
#pragma unroll
      for (int rr = 0; rr < 8; ++rr) {
        int c = mb * 16 + rr + 8 * half;
        outs[s][(((long)b * 64 + c) << 14) + pix] = acc[s][mb][rr] + bs[s][c];
      }
}

// ---------------------------------------------------------------------------
// off1: 1x1 conv over concat([crs, fused]) (K=128) + BN + SiLU, CO=64.
// Reads the two sources directly -> no materialized concat.
// ---------------------------------------------------------------------------
__global__ __launch_bounds__(256) void off1_wmma(
    const float* __restrict__ a0, const float* __restrict__ a1,
    const float* __restrict__ w, const float* __restrict__ bnp,
    float* __restrict__ out) {
  __shared__ unsigned lds_a[128][16];
  __shared__ unsigned lds_w[64][16];
  __shared__ float bninv[64], bnsh[64];
  const int tid = threadIdx.x;
  const int lane = tid & 31, wave = tid >> 5;
  const int half = lane >> 4, lanelo = lane & 15;
  const int p0 = blockIdx.x * 128;

  if (tid < 64) {
    float g = bnp[tid], bb = bnp[64 + tid], m = bnp[128 + tid], vv = bnp[192 + tid];
    float inv = g * rsqrtf(vv + BN_EPS);
    bninv[tid] = inv;
    bnsh[tid] = bb - m * inv;
  }

  v8f acc[4];
#pragma unroll
  for (int i = 0; i < 4; ++i) acc[i] = vzero8();

  for (int cc0 = 0; cc0 < 128; cc0 += 32) {
    __syncthreads();
    for (int u = tid; u < 128 * 16; u += 256) {
      int ml = u & 127, ku = u >> 7;
      int ci = cc0 + 2 * ku;                 // even; pair never straddles 63/64
      int p = p0 + ml;
      int b = p >> 14, pix = p & 16383;
      const float* src = (ci < 64)
          ? a0 + (((long)b * 64 + ci) << 14) + pix
          : a1 + (((long)b * 64 + (ci - 64)) << 14) + pix;
      lds_a[ml][ku] = pack2bf(src[0], src[16384]);
    }
    for (int u = tid; u < 64 * 16; u += 256) {
      int n = u & 63, ku = u >> 6;
      int ci = cc0 + 2 * ku;
      const float* wp = w + n * 128 + ci;
      lds_w[n][ku] = pack2bf(wp[0], wp[1]);
    }
    __syncthreads();
    Frag bf;
    load_frag(lds_a[wave * 16 + lanelo], half, bf);
#pragma unroll
    for (int mb = 0; mb < 4; ++mb) {
      Frag af;
      load_frag(lds_w[mb * 16 + lanelo], half, af);
      acc[mb] = wmma_bf16(af.v, bf.v, acc[mb]);
    }
  }

  int p = p0 + wave * 16 + lanelo;
  int b = p >> 14, pix = p & 16383;
#pragma unroll
  for (int mb = 0; mb < 4; ++mb)
#pragma unroll
    for (int rr = 0; rr < 8; ++rr) {
      int c = mb * 16 + rr + 8 * half;
      float v = acc[mb][rr] * bninv[c] + bnsh[c];
      out[(((long)b * 64 + c) << 14) + pix] = silu(v);
    }
}

// ---------------------------------------------------------------------------
// Bilinear upsample 64x64 -> 128x128, align_corners=True.
// ---------------------------------------------------------------------------
__global__ void upsample_ac(const float* __restrict__ s64, float* __restrict__ sem) {
  int idx = blockIdx.x * 256 + threadIdx.x;   // 8*64*128*128 threads
  int x = idx & 127, y = (idx >> 7) & 127;
  int pc = idx >> 14;                         // plane = b*64 + c
  float fy = y * (63.f / 127.f), fx = x * (63.f / 127.f);
  int y0 = (int)fy, x0 = (int)fx;
  int y1 = min(y0 + 1, 63), x1 = min(x0 + 1, 63);
  float wy = fy - y0, wx = fx - x0;
  const float* p = s64 + (long)pc * 4096;
  float v00 = p[y0 * 64 + x0], v01 = p[y0 * 64 + x1];
  float v10 = p[y1 * 64 + x0], v11 = p[y1 * 64 + x1];
  sem[idx] = (v00 * (1.f - wx) + v01 * wx) * (1.f - wy) +
             (v10 * (1.f - wx) + v11 * wx) * wy;
}

// ---------------------------------------------------------------------------
// 128-point radix-2 FFT in LDS; 64 worker threads per row (each owns i, i+64).
// sign = -1 forward, +1 inverse (scaling applied by callers).
// ---------------------------------------------------------------------------
__device__ __forceinline__ void fft128_lds(float* re, float* im, int t2, float sign) {
#pragma unroll
  for (int j = 0; j < 2; ++j) {
    int i = t2 + 64 * j;
    int br = __brev((unsigned)i) >> 25;
    if (br > i) {
      float tr = re[i]; re[i] = re[br]; re[br] = tr;
      float ti = im[i]; im[i] = im[br]; im[br] = ti;
    }
  }
  __syncthreads();
#pragma unroll
  for (int s = 0; s < 7; ++s) {
    int hm = 1 << s;
    int j = t2 & (hm - 1);
    int i0 = ((t2 >> s) << (s + 1)) + j;
    int i1 = i0 + hm;
    float ang = sign * 6.2831853071795864769f * (float)j / (float)(hm << 1);
    float wr, wi;
    __sincosf(ang, &wi, &wr);
    float ar = re[i0], ai = im[i0];
    float cr = re[i1], ci = im[i1];
    float vr = cr * wr - ci * wi;
    float vi = cr * wi + ci * wr;
    re[i0] = ar + vr; im[i0] = ai + vi;
    re[i1] = ar - vr; im[i1] = ai - vi;
    __syncthreads();
  }
}

// Row-wise forward FFT of real x2 -> complex buffer. 4 rows per block.
__global__ __launch_bounds__(256) void fft_rows_fwd(const float* __restrict__ x2,
                                                    float2* __restrict__ cb) {
  __shared__ float re[4][128], im[4][128];
  int tid = threadIdx.x, slot = tid >> 6, t2 = tid & 63;
  long row = (long)blockIdx.x * 4 + slot;     // row = plane*128 + y
  const float* src = x2 + row * 128;
  re[slot][t2] = src[t2];       im[slot][t2] = 0.f;
  re[slot][t2 + 64] = src[t2 + 64]; im[slot][t2 + 64] = 0.f;
  __syncthreads();
  fft128_lds(re[slot], im[slot], t2, -1.f);
  float2* dst = cb + row * 128;
  dst[t2]      = make_float2(re[slot][t2], im[slot][t2]);
  dst[t2 + 64] = make_float2(re[slot][t2 + 64], im[slot][t2 + 64]);
}

// Column pass: fwd col FFT, pointwise * x1, inverse col FFT (1/128 scale).
__global__ __launch_bounds__(256) void fft_cols(float2* __restrict__ cb,
                                                const float* __restrict__ x1v) {
  __shared__ float re[4][128], im[4][128];
  int tid = threadIdx.x, slot = tid >> 6, t2 = tid & 63;
  int plane = blockIdx.x >> 5;                // 32 column-groups of 4 per plane
  int v0 = (blockIdx.x & 31) * 4;
  float2* base = cb + (long)plane * 16384;
  for (int e = tid; e < 512; e += 256) {
    int u = e >> 2, v = e & 3;
    float2 val = base[u * 128 + v0 + v];
    re[v][u] = val.x; im[v][u] = val.y;
  }
  __syncthreads();
  fft128_lds(re[slot], im[slot], t2, -1.f);
  const float* xp = x1v + (long)plane * 16384 + (v0 + slot);
  {
    float a0 = xp[t2 * 128];
    re[slot][t2] *= a0; im[slot][t2] *= a0;
    float a1 = xp[(t2 + 64) * 128];
    re[slot][t2 + 64] *= a1; im[slot][t2 + 64] *= a1;
  }
  __syncthreads();
  fft128_lds(re[slot], im[slot], t2, +1.f);
  const float sc = 1.f / 128.f;
  for (int e = tid; e < 512; e += 256) {
    int u = e >> 2, v = e & 3;
    base[u * 128 + v0 + v] = make_float2(re[v][u] * sc, im[v][u] * sc);
  }
}

// Row inverse FFT + |.| + enh/gate blend epilogue -> fused.
__global__ __launch_bounds__(256) void fft_rows_inv_epi(
    const float2* __restrict__ cb, const float* __restrict__ sem,
    const float* __restrict__ glog, const float* __restrict__ alpha,
    const float* __restrict__ beta, float* __restrict__ fused) {
  __shared__ float re[4][128], im[4][128];
  int tid = threadIdx.x, slot = tid >> 6, t2 = tid & 63;
  long row = (long)blockIdx.x * 4 + slot;
  int c = (int)((row >> 7) & 63);
  const float2* src = cb + row * 128;
  float2 a0 = src[t2], a1 = src[t2 + 64];
  re[slot][t2] = a0.x;       im[slot][t2] = a0.y;
  re[slot][t2 + 64] = a1.x;  im[slot][t2 + 64] = a1.y;
  __syncthreads();
  fft128_lds(re[slot], im[slot], t2, +1.f);
  const float sc = 1.f / 128.f;
  float al = alpha[c], be = beta[c];
  long base = row * 128;
#pragma unroll
  for (int j = 0; j < 2; ++j) {
    int i = t2 + 64 * j;
    float fr = re[slot][i] * sc, fi = im[slot][i] * sc;
    float mag = sqrtf(fr * fr + fi * fi);
    float s = sem[base + i];
    float enh = mag * al + s * be;
    float g = 1.f / (1.f + __expf(-glog[base + i]));
    fused[base + i] = s * (1.f - g) + enh * g;
  }
}

// ---------------------------------------------------------------------------
// grid_sample (align_corners, zeros padding) warps + attention combine.
// ---------------------------------------------------------------------------
struct Samp {
  int i0, i1, i2, i3;
  float w0, w1, w2, w3;
};

__device__ __forceinline__ Samp mk_samp(float bx, float by, float ox, float oy) {
  float px = (bx + ox * (1.f / 128.f) + 1.f) * 0.5f * 127.f;
  float py = (by + oy * (1.f / 128.f) + 1.f) * 0.5f * 127.f;
  float x0f = floorf(px), y0f = floorf(py);
  float wx = px - x0f, wy = py - y0f;
  int x0 = (int)x0f, y0 = (int)y0f, x1 = x0 + 1, y1 = y0 + 1;
  bool vx0 = (x0 >= 0) && (x0 <= 127), vx1 = (x1 >= 0) && (x1 <= 127);
  bool vy0 = (y0 >= 0) && (y0 <= 127), vy1 = (y1 >= 0) && (y1 <= 127);
  int cx0 = min(max(x0, 0), 127), cx1 = min(max(x1, 0), 127);
  int cy0 = min(max(y0, 0), 127), cy1 = min(max(y1, 0), 127);
  Samp s;
  s.i0 = cy0 * 128 + cx0; s.w0 = (vx0 && vy0) ? (1.f - wx) * (1.f - wy) : 0.f;
  s.i1 = cy0 * 128 + cx1; s.w1 = (vx1 && vy0) ? wx * (1.f - wy) : 0.f;
  s.i2 = cy1 * 128 + cx0; s.w2 = (vx0 && vy1) ? (1.f - wx) * wy : 0.f;
  s.i3 = cy1 * 128 + cx1; s.w3 = (vx1 && vy1) ? wx * wy : 0.f;
  return s;
}

__global__ void warp_combine(const float* __restrict__ crs,
                             const float* __restrict__ fused,
                             const float* __restrict__ res,
                             float* __restrict__ out) {
  int p = blockIdx.x * 256 + threadIdx.x;    // 8*128*128 threads
  int b = p >> 14, pix = p & 16383;
  int y = pix >> 7, x = pix & 127;
  float bx = -1.f + x * (2.f / 127.f);
  float by = -1.f + y * (2.f / 127.f);
  const float* rb = res + (long)b * 10 * 16384 + pix;
  float rch[10];
#pragma unroll
  for (int i = 0; i < 10; ++i) rch[i] = rb[(long)i * 16384];
  Samp sl0 = mk_samp(bx, by, rch[0], rch[1]);
  Samp sl1 = mk_samp(bx, by, rch[2], rch[3]);
  Samp sh0 = mk_samp(bx, by, rch[4], rch[5]);
  Samp sh1 = mk_samp(bx, by, rch[6], rch[7]);
  float att0 = 1.f + tanhf(rch[8]);
  float att1 = 1.f + tanhf(rch[9]);
  for (int c = 0; c < 64; ++c) {
    Samp L = (c < 32) ? sl0 : sl1;
    Samp Hh = (c < 32) ? sh0 : sh1;
    const float* cp = crs + (((long)b * 64 + c) << 14);
    const float* fp = fused + (((long)b * 64 + c) << 14);
    float cs = cp[L.i0] * L.w0 + cp[L.i1] * L.w1 + cp[L.i2] * L.w2 + cp[L.i3] * L.w3;
    float fs = fp[Hh.i0] * Hh.w0 + fp[Hh.i1] * Hh.w1 + fp[Hh.i2] * Hh.w2 + fp[Hh.i3] * Hh.w3;
    out[(((long)b * 64 + c) << 14) + pix] = fs * att0 + cs * att1;
  }
}

// ---------------------------------------------------------------------------
// Host-side orchestration.
// ---------------------------------------------------------------------------
extern "C" void kernel_launch(void* const* d_in, const int* in_sizes, int n_in,
                              void* d_out, int out_size, void* d_ws, size_t ws_size,
                              hipStream_t stream) {
  (void)in_sizes; (void)n_in; (void)out_size; (void)ws_size;
  const float* coarse   = (const float*)d_in[0];
  const float* semantic = (const float*)d_in[1];
  const float* w_sp     = (const float*)d_in[2];
  const float* bn_sp    = (const float*)d_in[3];
  const float* w_se     = (const float*)d_in[4];
  const float* bn_se    = (const float*)d_in[5];
  const float* w_dw1    = (const float*)d_in[6];
  const float* b_dw1    = (const float*)d_in[7];
  const float* w_dw2    = (const float*)d_in[8];
  const float* b_dw2    = (const float*)d_in[9];
  const float* alpha    = (const float*)d_in[10];
  const float* beta     = (const float*)d_in[11];
  const float* w_gate   = (const float*)d_in[12];
  const float* b_gate   = (const float*)d_in[13];
  const float* w_off1   = (const float*)d_in[14];
  const float* bn_off   = (const float*)d_in[15];
  const float* w_off2   = (const float*)d_in[16];
  float* out = (float*)d_out;

  char* ws = (char*)d_ws;
  size_t off = 0;
  auto carve = [&](size_t bytes) -> void* {
    void* p = ws + off;
    off += (bytes + 255) & ~(size_t)255;
    return p;
  };
  const size_t PLANE = 8ull * 64 * 128 * 128;          // 8.39M elems
  float*  sem64 = (float*)carve(8ull * 64 * 64 * 64 * 4);
  float*  sem   = (float*)carve(PLANE * 4);
  float*  x1    = (float*)carve(PLANE * 4);
  float*  x2    = (float*)carve(PLANE * 4);
  float*  gl    = (float*)carve(PLANE * 4);
  float2* cb    = (float2*)carve(PLANE * 8);
  float*  fusedb= (float*)carve(PLANE * 4);
  // Aliased buffers (lifetimes disjoint): x2 dead after fft_rows_fwd,
  // x1 dead after fft_cols, gl dead after fft_rows_inv_epi.
  float* crs  = x2;
  float* offp = x1;
  float* resb = gl;

  // 1. semantic branch conv3x3 (128->64 @ 64x64) + BN + SiLU
  conv3x3_wmma<128, 64, 64, 64, 64, 1><<<256, 256, 0, stream>>>(
      semantic, w_se, bn_se, sem64);
  // 2. align-corners bilinear upsample 64 -> 128
  upsample_ac<<<32768, 256, 0, stream>>>(sem64, sem);
  // 3. fused 1x1 convs: x1, x2, gate logits
  ffm_pre_wmma<<<1024, 256, 0, stream>>>(sem, w_dw1, b_dw1, w_dw2, b_dw2,
                                         w_gate, b_gate, x1, x2, gl);
  // 4-6. FFM frequency enhancer: fft2(x2), * x1, ifft2, |.|, blend
  fft_rows_fwd<<<16384, 256, 0, stream>>>(x2, cb);
  fft_cols<<<16384, 256, 0, stream>>>(cb, x1);
  fft_rows_inv_epi<<<16384, 256, 0, stream>>>(cb, sem, gl, alpha, beta, fusedb);
  // 7. spatial branch conv3x3 (64->64 @ 128x128) + BN + SiLU
  conv3x3_wmma<64, 64, 64, 128, 128, 1><<<1024, 256, 0, stream>>>(
      coarse, w_sp, bn_sp, crs);
  // 8. offset predictor stage 1: 1x1 conv over concat (K=128) + BN + SiLU
  off1_wmma<<<1024, 256, 0, stream>>>(crs, fusedb, w_off1, bn_off, offp);
  // 9. offset predictor stage 2: 3x3 conv 64 -> 10 (M padded to 16)
  conv3x3_wmma<64, 16, 10, 128, 128, 0><<<1024, 256, 0, stream>>>(
      offp, w_off2, nullptr, resb);
  // 10. dual grid_sample warps + attention combine
  warp_combine<<<512, 256, 0, stream>>>(crs, fusedb, resb, out);
}
